// TokenFlow_45913200394474
// MI455X (gfx1250) — matrix operand order repaired and checked by reading, not compiled, not run on hardware
//
#include <hip/hip_runtime.h>
#include <stdint.h>

// ---- problem constants ----
#define NFR     4
#define NHEADS  8
#define SEQ     576
#define DIMN    640
#define HD      80          // head dim
#define HDP     96          // padded head dim (3 x 32 for WMMA K chunks)
#define MTOT    (12 * SEQ)  // 6912 rows across all frames

typedef __attribute__((ext_vector_type(16))) __bf16 v16bf;
typedef __attribute__((ext_vector_type(8)))  float  v8f;

union FragAB { v16bf v; uint4 q[2]; unsigned short u[16]; };
union FragC  { v8f   v; float f[8]; };

__device__ __forceinline__ unsigned short f2bf(float x) {
    union { float f; unsigned u; } c; c.f = x;
    unsigned u = c.u + 0x7FFFu + ((c.u >> 16) & 1u);   // round-to-nearest-even
    return (unsigned short)(u >> 16);
}

__device__ __forceinline__ float rmax16(float v) {
#pragma unroll
    for (int m = 1; m < 16; m <<= 1) v = fmaxf(v, __shfl_xor(v, m, 16));
    return v;
}
__device__ __forceinline__ float rsum16(float v) {
#pragma unroll
    for (int m = 1; m < 16; m <<= 1) v += __shfl_xor(v, m, 16);
    return v;
}

// async global -> LDS (CDNA5), 16 bytes per lane; tracked by ASYNCcnt
__device__ __forceinline__ void async_b128(const void* gptr, void* lds_ptr) {
    unsigned lds_off = (unsigned)(uintptr_t)lds_ptr;  // low 32 bits = LDS offset
    asm volatile("global_load_async_to_lds_b128 %0, %1, off"
                 :: "v"(lds_off), "v"(gptr) : "memory");
}
__device__ __forceinline__ void wait_async0() {
    asm volatile("s_wait_asynccnt 0x0" ::: "memory");
}

// ---------------------------------------------------------------------------
// elementwise helpers
// ---------------------------------------------------------------------------
__global__ void tf_zero_u16(unsigned short* p, int n) {
    int i = blockIdx.x * blockDim.x + threadIdx.x;
    int s = gridDim.x * blockDim.x;
    for (; i < n; i += s) p[i] = 0;
}

__global__ void tf_cvt_bf16(const float* __restrict__ src,
                            unsigned short* __restrict__ dst, int n) {
    int i = blockIdx.x * blockDim.x + threadIdx.x;
    int s = gridDim.x * blockDim.x;
    for (; i < n; i += s) dst[i] = f2bf(src[i]);
}

// ---------------------------------------------------------------------------
// GEMM: C[M,640] = A[M,640](bf16) @ W[640,640](bf16)^T
// block = 128 threads (4 waves), tile 128x64, BK=32.
// Async global->LDS double-buffered pipeline.
// mode 0/1: write q/k layout  [frame, head, s, 96] bf16
// mode 2  : write v  layout   [frame, head, d, s ] bf16 (transposed)
// mode 3  : write f32 out [m, 640] + bias
// ---------------------------------------------------------------------------
__global__ __launch_bounds__(128)
void tf_gemm_wmma(const unsigned short* __restrict__ A,
                  const unsigned short* __restrict__ W,
                  void* __restrict__ dst,
                  const float* __restrict__ bias,
                  int mode) {
    __shared__ uint4 sA[2][512];   // 128 rows x 32 bf16 per buffer
    __shared__ uint4 sB[2][256];   //  64 rows x 32 bf16 per buffer

    const int tid  = threadIdx.x;
    const int lane = tid & 31;
    const int wav  = tid >> 5;       // 0..3 -> N sub-tile
    const int lo   = lane & 15;
    const int hi   = lane >> 4;
    const int mBase = blockIdx.x * 128;
    const int nBase = blockIdx.y * 64;

    // issue async loads for one K-tile into buffer `buf`
    auto prefetch = [&](int buf, int k0) {
#pragma unroll
        for (int it = 0; it < 4; ++it) {                 // A: 512 b128 chunks
            int chunk = tid + it * 128;
            int row = chunk >> 2;
            int col = (chunk & 3) * 8;
            async_b128(A + (size_t)(mBase + row) * DIMN + k0 + col, &sA[buf][chunk]);
        }
#pragma unroll
        for (int it = 0; it < 2; ++it) {                 // B: 256 b128 chunks
            int chunk = tid + it * 128;
            int row = chunk >> 2;
            int col = (chunk & 3) * 8;
            async_b128(W + (size_t)(nBase + row) * DIMN + k0 + col, &sB[buf][chunk]);
        }
    };

    FragC acc[8];
#pragma unroll
    for (int t = 0; t < 8; ++t)
#pragma unroll
        for (int r = 0; r < 8; ++r) acc[t].f[r] = 0.0f;

    prefetch(0, 0);
    int cur = 0;
    for (int k0 = 0; k0 < DIMN; k0 += 32) {
        wait_async0();          // my async writes into buf `cur` done
        __syncthreads();        // everyone's done -> buf `cur` valid, prev reads retired
        if (k0 + 32 < DIMN) prefetch(cur ^ 1, k0 + 32);

        const unsigned short* pA = (const unsigned short*)sA[cur];
        const unsigned short* pB = (const unsigned short*)sB[cur];

        FragAB fb;  // B fragment: N = nBase + wav*16 + lo, two 8-halfword K runs
        fb.q[0] = *(const uint4*)(pB + (wav * 16 + lo) * 32 + hi * 8);
        fb.q[1] = *(const uint4*)(pB + (wav * 16 + lo) * 32 + hi * 8 + 16);

#pragma unroll
        for (int mt = 0; mt < 8; ++mt) {
            FragAB fa;  // A fragment: M = mt*16 + lo, two 8-halfword K runs
            int base = (mt * 16 + lo) * 32 + hi * 8;
            fa.q[0] = *(const uint4*)(pA + base);
            fa.q[1] = *(const uint4*)(pA + base + 16);
            acc[mt].v = __builtin_amdgcn_wmma_f32_16x16x32_bf16(
                false, fa.v, false, fb.v, (short)0, acc[mt].v, false, false);
        }
        cur ^= 1;
    }

    // store: C layout -> lane holds rows m = r + hi*8, col n = nBase+wav*16+lo
    const int n = nBase + wav * 16 + lo;
#pragma unroll
    for (int mt = 0; mt < 8; ++mt) {
#pragma unroll
        for (int r = 0; r < 8; ++r) {
            int m = mBase + mt * 16 + r + hi * 8;
            float val = acc[mt].f[r];
            if (mode == 3) {
                ((float*)dst)[(size_t)m * DIMN + n] = val + bias[n];
            } else {
                int frame = m / SEQ, s = m % SEQ;
                int head = n / HD, d = n % HD;
                unsigned short bv = f2bf(val);
                if (mode == 2) {   // V transposed: [frame, head, d, s]
                    ((unsigned short*)dst)[(((size_t)frame * NHEADS + head) * HD + d) * SEQ + s] = bv;
                } else {           // Q/K: [frame, head, s, 96]
                    ((unsigned short*)dst)[(((size_t)frame * NHEADS + head) * SEQ + s) * HDP + d] = bv;
                }
            }
        }
    }
}

// ---------------------------------------------------------------------------
// Flash attention: one wave per (frame, head, 32-query pair-tile).
// Two 16-row query tiles per wave so每 K/V fragment feeds two WMMAs.
// qb/kb: [12, 8, 576, 96] bf16 (padded, zeros in 80..95)
// vt   : [12, 8, 80, 576] bf16 (transposed)
// ob   : [12, 576, 640]   bf16 (heads merged)
// ---------------------------------------------------------------------------
__global__ __launch_bounds__(128)
void tf_attn_wmma(const unsigned short* __restrict__ qb,
                  const unsigned short* __restrict__ kb,
                  const unsigned short* __restrict__ vt,
                  unsigned short* __restrict__ ob) {
    __shared__ uint4 psh[4 * 128];           // per-wave: two 16x32 bf16 P tiles
    const int lane = threadIdx.x & 31;
    const int wav  = threadIdx.x >> 5;
    const int lo = lane & 15, hi = lane >> 4;
    unsigned short* pb = (unsigned short*)(psh + wav * 128);

    const int NQ2 = SEQ / 32;                // 18 query pair-tiles
    const int wid = blockIdx.x * 4 + wav;    // 12*8*18 = 1728 waves
    if (wid >= 12 * NHEADS * NQ2) return;
    const int frame = wid / (NHEADS * NQ2);
    const int head  = (wid / NQ2) % NHEADS;
    const int qt2   = wid % NQ2;

    int kvf0, nkv;
    if (frame < NFR)            { kvf0 = frame; nkv = SEQ; }        // self
    else if (frame < 2 * NFR)   { kvf0 = NFR;   nkv = NFR * SEQ; }  // uncond
    else                        { kvf0 = 2*NFR; nkv = NFR * SEQ; }  // cond
    const float scale = 0.11180339887498949f; // 1/sqrt(80)

    // Q fragments: 2 query tiles x 3 K-chunks over padded head dim
    FragAB aq[2][3];
#pragma unroll
    for (int qi = 0; qi < 2; ++qi) {
        const unsigned short* qrow =
            qb + (((size_t)frame * NHEADS + head) * SEQ + qt2 * 32 + qi * 16 + lo) * HDP;
#pragma unroll
        for (int c = 0; c < 3; ++c) {
            int d0 = c * 32 + hi * 8;
            aq[qi][c].q[0] = *(const uint4*)(qrow + d0);
            aq[qi][c].q[1] = *(const uint4*)(qrow + d0 + 16);
        }
    }

    FragC o[2][5];
    float m_run[2][8], l_run[2][8];
#pragma unroll
    for (int qi = 0; qi < 2; ++qi) {
#pragma unroll
        for (int t = 0; t < 5; ++t)
#pragma unroll
            for (int r = 0; r < 8; ++r) o[qi][t].f[r] = 0.0f;
#pragma unroll
        for (int r = 0; r < 8; ++r) { m_run[qi][r] = -1e30f; l_run[qi][r] = 0.0f; }
    }

    for (int kb0 = 0; kb0 < nkv; kb0 += 32) {
        const int kf = kvf0 + kb0 / SEQ;
        const int ks = kb0 % SEQ;          // 576 % 32 == 0: no frame straddle

        // ---- logits: S[qi] = Q[qi] K^T; each K fragment used twice ----
        FragC s[2][2];
#pragma unroll
        for (int qi = 0; qi < 2; ++qi)
#pragma unroll
            for (int t = 0; t < 2; ++t)
#pragma unroll
                for (int r = 0; r < 8; ++r) s[qi][t].f[r] = 0.0f;

#pragma unroll
        for (int t = 0; t < 2; ++t) {
            const unsigned short* krow =
                kb + (((size_t)kf * NHEADS + head) * SEQ + ks + t * 16 + lo) * HDP;
#pragma unroll
            for (int c = 0; c < 3; ++c) {
                FragAB fk;   // B: N = key (t*16+lo), K = hd chunk c, two 8-element runs
                int off = c * 32 + hi * 8;
                fk.q[0] = *(const uint4*)(krow + off);
                fk.q[1] = *(const uint4*)(krow + off + 16);
                s[0][t].v = __builtin_amdgcn_wmma_f32_16x16x32_bf16(
                    false, aq[0][c].v, false, fk.v, (short)0, s[0][t].v, false, false);
                s[1][t].v = __builtin_amdgcn_wmma_f32_16x16x32_bf16(
                    false, aq[1][c].v, false, fk.v, (short)0, s[1][t].v, false, false);
            }
        }

        // ---- online softmax per query tile; P -> LDS (A-layout staging) ----
#pragma unroll
        for (int qi = 0; qi < 2; ++qi) {
#pragma unroll
            for (int r = 0; r < 8; ++r) {
                float v0 = s[qi][0].f[r] * scale;
                float v1 = s[qi][1].f[r] * scale;
                float bm = rmax16(fmaxf(v0, v1));
                float mn = fmaxf(m_run[qi][r], bm);
                float corr = __expf(m_run[qi][r] - mn);
                float p0 = __expf(v0 - mn);
                float p1 = __expf(v1 - mn);
                float rs = rsum16(p0 + p1);
                l_run[qi][r] = l_run[qi][r] * corr + rs;
                m_run[qi][r] = mn;
#pragma unroll
                for (int t = 0; t < 5; ++t) o[qi][t].f[r] *= corr;
                int m = r + hi * 8;
                pb[qi * 512 + m * 32 + lo]      = f2bf(p0);
                pb[qi * 512 + m * 32 + 16 + lo] = f2bf(p1);
            }
        }
        asm volatile("s_wait_dscnt 0x0" ::: "memory");
        FragAB ap[2];
#pragma unroll
        for (int qi = 0; qi < 2; ++qi) {
            ap[qi].q[0] = *(const uint4*)(pb + qi * 512 + lo * 32 + hi * 8);
            ap[qi].q[1] = *(const uint4*)(pb + qi * 512 + lo * 32 + hi * 8 + 16);
        }

        // ---- O[qi] += P[qi] V; each V fragment used twice ----
#pragma unroll
        for (int t = 0; t < 5; ++t) {
            const unsigned short* vrow =
                vt + (((size_t)kf * NHEADS + head) * HD + t * 16 + lo) * SEQ + ks + hi * 8;
            FragAB fv;   // B: N = hd col, K = two 8-key runs per lane half
            fv.q[0] = *(const uint4*)(vrow);
            fv.q[1] = *(const uint4*)(vrow + 16);
            o[0][t].v = __builtin_amdgcn_wmma_f32_16x16x32_bf16(
                false, ap[0].v, false, fv.v, (short)0, o[0][t].v, false, false);
            o[1][t].v = __builtin_amdgcn_wmma_f32_16x16x32_bf16(
                false, ap[1].v, false, fv.v, (short)0, o[1][t].v, false, false);
        }
    }

    // ---- normalize + store merged-head bf16 ----
#pragma unroll
    for (int qi = 0; qi < 2; ++qi)
#pragma unroll
        for (int t = 0; t < 5; ++t)
#pragma unroll
            for (int r = 0; r < 8; ++r) {
                int m = r + hi * 8;
                int s = qt2 * 32 + qi * 16 + m;
                int col = head * HD + t * 16 + lo;
                ob[((size_t)frame * SEQ + s) * DIMN + col] =
                    f2bf(o[qi][t].f[r] / l_run[qi][r]);
            }
}

// ---------------------------------------------------------------------------
extern "C" void kernel_launch(void* const* d_in, const int* in_sizes, int n_in,
                              void* d_out, int out_size, void* d_ws, size_t ws_size,
                              hipStream_t stream) {
    const float* x  = (const float*)d_in[0];
    const float* Wq = (const float*)d_in[1];
    const float* Wk = (const float*)d_in[2];
    const float* Wv = (const float*)d_in[3];
    const float* Wo = (const float*)d_in[4];
    const float* bo = (const float*)d_in[5];

    // workspace layout (ushort element offsets; all 16B aligned)
    unsigned short* ws = (unsigned short*)d_ws;
    const size_t nX  = (size_t)MTOT * DIMN;             // 4,423,680
    const size_t nW  = (size_t)DIMN * DIMN;             //   409,600
    const size_t nQK = (size_t)12 * NHEADS * SEQ * HDP; // 5,308,416 (padded)
    unsigned short* xb   = ws;
    unsigned short* Wqb  = xb  + nX;
    unsigned short* Wkb  = Wqb + nW;
    unsigned short* Wvb  = Wkb + nW;
    unsigned short* Wob  = Wvb + nW;
    unsigned short* qbf  = Wob + nW;
    unsigned short* kbf  = qbf + nQK;
    unsigned short* vtb  = kbf + nQK;                   // [12,8,80,576]
    unsigned short* attb = vtb + nX;                    // [12,576,640]

    // 1) zero the padded q/k buffers (pad cols 80..95 must be 0)
    tf_zero_u16<<<1024, 256, 0, stream>>>(qbf, (int)nQK);
    tf_zero_u16<<<1024, 256, 0, stream>>>(kbf, (int)nQK);

    // 2) f32 -> bf16 conversions
    tf_cvt_bf16<<<2048, 256, 0, stream>>>(x,  xb,  (int)nX);
    tf_cvt_bf16<<<512,  256, 0, stream>>>(Wq, Wqb, (int)nW);
    tf_cvt_bf16<<<512,  256, 0, stream>>>(Wk, Wkb, (int)nW);
    tf_cvt_bf16<<<512,  256, 0, stream>>>(Wv, Wvb, (int)nW);
    tf_cvt_bf16<<<512,  256, 0, stream>>>(Wo, Wob, (int)nW);

    // 3) QKV projections (WMMA GEMMs, async double-buffered)
    dim3 gg(MTOT / 128, DIMN / 64);
    tf_gemm_wmma<<<gg, 128, 0, stream>>>(xb, Wqb, qbf, nullptr, 0);
    tf_gemm_wmma<<<gg, 128, 0, stream>>>(xb, Wkb, kbf, nullptr, 1);
    tf_gemm_wmma<<<gg, 128, 0, stream>>>(xb, Wvb, vtb, nullptr, 2);

    // 4) attention (flash-style, WMMA logits + PV, 2 query tiles / wave)
    tf_attn_wmma<<<(12 * NHEADS * (SEQ / 32)) / 4, 128, 0, stream>>>(qbf, kbf, vtb, attb);

    // 5) output projection + bias -> f32
    tf_gemm_wmma<<<gg, 128, 0, stream>>>(attb, Wob, d_out, bo, 3);
}
